// ExtendedISAModule_23347442221064
// MI455X (gfx1250) — compile-verified, matrix operand order checked by reading, not run on hardware
//
#include <hip/hip_runtime.h>

typedef __attribute__((ext_vector_type(16))) _Float16 v16h;
typedef __attribute__((ext_vector_type(8)))  _Float16 v8h;
typedef __attribute__((ext_vector_type(4)))  _Float16 v4h;
typedef __attribute__((ext_vector_type(8)))  float    v8f;
typedef __attribute__((ext_vector_type(4)))  int      v4i;

#define TM 128
#define TN 128
#define TK 32
#define LDSP 40   // LDS row pitch in halves: 80B -> 16B-aligned b128, conflict-free frag reads

#if defined(__gfx1250__) && __has_builtin(__builtin_amdgcn_global_load_async_to_lds_b128)
#define USE_ASYNC_LDS 1
#else
#define USE_ASYNC_LDS 0
#endif

union Frag  { v16h v; v8h h[2]; };
union Stage { _Float16 e[16]; v8h h[2]; };

// C = alpha * sum_h opA(A_h)[M,K] * B_h[K,N] (+resid) (+bias[m]) (relu?)
// A,B f16; fp32 WMMA accumulation; writes C32 (f32) and/or C16 (f16).
// blockIdx.z batches heads (strideCh); nHsum>1 sums heads inside the K loop.
// Double-buffered pipeline: async global->LDS for A, register-staged transpose for B.
__global__ __launch_bounds__(256)
void gemm_wmma_f16(const _Float16* __restrict__ A, int lda, int strideAh, int transA,
                   const _Float16* __restrict__ B, int ldb, int strideBh,
                   float* __restrict__ C32, _Float16* __restrict__ C16,
                   int ldc, int strideCh,
                   const float* __restrict__ resid, const float* __restrict__ bias,
                   int K, int nHsum, float alpha, int relu)
{
    __shared__ __align__(16) _Float16 As[2 * TM * LDSP];
    __shared__ __align__(16) _Float16 Bs[2 * TN * LDSP];

    const int t    = threadIdx.x;
    const int lane = t & 31;        // wave32
    const int w    = t >> 5;        // 8 waves
    const int wr   = w >> 2;        // 0..1 -> 64 rows
    const int wc   = w & 3;         // 0..3 -> 32 cols
    const int half = lane >> 4;
    const int l16  = lane & 15;

    const int Nbase = blockIdx.x * TN;
    const int Mbase = blockIdx.y * TM;
    const int z     = blockIdx.z;

    // async A staging map (non-trans): thread copies 2x16B of row aR at col aC
    const int aR = t >> 1;
    const int aC = (t & 1) * 16;
    // register staging map (B, and A when transposed): column bN, 16 contiguous k rows
    const int bN  = t & 127;
    const int bK0 = (t >> 7) * 16;

    const v8f vzero = {0.f, 0.f, 0.f, 0.f, 0.f, 0.f, 0.f, 0.f};
    v8f acc[4][2];
#pragma unroll
    for (int i = 0; i < 4; ++i)
#pragma unroll
        for (int j = 0; j < 2; ++j) acc[i][j] = vzero;

    // staging state: (head, k0) advanced incrementally
    const _Float16* aH = A + z * strideAh + (transA ? Mbase : (Mbase + aR) * lda + aC);
    const _Float16* bH = B + z * strideBh + Nbase;
    int k0n = 0;
    const int steps = nHsum * (K / TK);

    Stage br, ar;

    auto issueA = [&](int buf) {
#if USE_ASYNC_LDS
        if (!transA) {
            __builtin_amdgcn_global_load_async_to_lds_b128(
                (__attribute__((address_space(1))) v4i*)(aH + k0n),
                (__attribute__((address_space(3))) v4i*)(&As[buf * TM * LDSP + aR * LDSP + aC]), 0, 0);
            __builtin_amdgcn_global_load_async_to_lds_b128(
                (__attribute__((address_space(1))) v4i*)(aH + k0n + 8),
                (__attribute__((address_space(3))) v4i*)(&As[buf * TM * LDSP + aR * LDSP + aC + 8]), 0, 0);
        }
#endif
    };
    auto loadStage = [&]() {
#pragma unroll
        for (int i = 0; i < 16; ++i)
            br.e[i] = bH[(k0n + bK0 + i) * ldb + bN];
        if (transA) {
#pragma unroll
            for (int i = 0; i < 16; ++i)
                ar.e[i] = aH[(k0n + bK0 + i) * lda + bN];
        }
#if !USE_ASYNC_LDS
        if (!transA) {
            ar.h[0] = *(const v8h*)(aH + k0n);
            ar.h[1] = *(const v8h*)(aH + k0n + 8);
        }
#endif
    };
    auto storeStage = [&](int buf) {
        *(v8h*)&Bs[buf * TN * LDSP + bN * LDSP + bK0]     = br.h[0];
        *(v8h*)&Bs[buf * TN * LDSP + bN * LDSP + bK0 + 8] = br.h[1];
        if (transA) {
            *(v8h*)&As[buf * TM * LDSP + bN * LDSP + bK0]     = ar.h[0];
            *(v8h*)&As[buf * TM * LDSP + bN * LDSP + bK0 + 8] = ar.h[1];
        }
#if !USE_ASYNC_LDS
        if (!transA) {
            *(v8h*)&As[buf * TM * LDSP + aR * LDSP + aC]     = ar.h[0];
            *(v8h*)&As[buf * TM * LDSP + aR * LDSP + aC + 8] = ar.h[1];
        }
#endif
    };
    auto waitAsync = [&]() {
#if USE_ASYNC_LDS
        if (!transA) {
#if __has_builtin(__builtin_amdgcn_s_wait_asynccnt)
            __builtin_amdgcn_s_wait_asynccnt(0);
#else
            asm volatile("s_wait_asynccnt 0" ::: "memory");
#endif
        }
#endif
    };
    auto advance = [&]() {
        k0n += TK;
        if (k0n == K) { k0n = 0; aH += strideAh; bH += strideBh; }
    };

    // ---- prologue: stage step 0 into buffer 0 ----
    issueA(0);
    loadStage();
    storeStage(0);
    waitAsync();
    advance();
    __syncthreads();

    for (int s = 0; s < steps; ++s) {
        const int cur = s & 1, nxt = cur ^ 1;
        const bool more = (s + 1 < steps);
        if (more) { issueA(nxt); loadStage(); }   // overlap with compute below

        // ---- compute from buffer cur ----
        // f16 A/B-frag element i <-> K = 16*(i>>3) + 8*half + (i&7):
        // two 16B chunks at k-offsets 8*half and 16+8*half.
        const _Float16* aF = &As[cur * TM * LDSP + (wr * 64 + l16) * LDSP + 8 * half];
        const _Float16* bF = &Bs[cur * TN * LDSP + (wc * 32 + l16) * LDSP + 8 * half];
        Frag bf[2];
#pragma unroll
        for (int fc = 0; fc < 2; ++fc) {
            const _Float16* p = bF + fc * 16 * LDSP;
            bf[fc].h[0] = *(const v8h*)(p);
            bf[fc].h[1] = *(const v8h*)(p + 16);
        }
#pragma unroll
        for (int fr = 0; fr < 4; ++fr) {
            Frag af;
            const _Float16* p = aF + fr * 16 * LDSP;
            af.h[0] = *(const v8h*)(p);
            af.h[1] = *(const v8h*)(p + 16);
#pragma unroll
            for (int fc = 0; fc < 2; ++fc) {
                acc[fr][fc] = __builtin_amdgcn_wmma_f32_16x16x32_f16(
                    false, af.v, false, bf[fc].v, (short)0, acc[fr][fc],
                    false, false);
            }
        }

        if (more) { storeStage(nxt); waitAsync(); advance(); }
        __syncthreads();
    }

    // ---- epilogue: lane -> n = l16, VGPR r -> m = r + 8*half ----
    const int cRow = Mbase + wr * 64 + half * 8;
    const int cCol = Nbase + wc * 32 + l16;
#pragma unroll
    for (int fr = 0; fr < 4; ++fr) {
#pragma unroll
        for (int fc = 0; fc < 2; ++fc) {
#pragma unroll
            for (int r = 0; r < 8; ++r) {
                int gm  = cRow + fr * 16 + r;
                int off = gm * ldc + cCol + fc * 16;
                float v = acc[fr][fc][r] * alpha;
                if (resid) v += resid[off];
                if (bias)  v += bias[gm];
                if (relu)  v = v > 0.f ? v : 0.f;
                if (C32) C32[z * strideCh + off] = v;
                if (C16) C16[z * strideCh + off] = (_Float16)v;
            }
        }
    }
}

// softmax over rows n per column m of S[h] (jax softmax axis=1 on [H,N,N]); emits f16 A
__global__ __launch_bounds__(256)
void softmax_cols(float* __restrict__ S, _Float16* __restrict__ A16, int n)
{
    const int m = blockIdx.x * 256 + threadIdx.x;
    float*    Sh = S   + (size_t)blockIdx.y * n * n;
    _Float16* Ah = A16 + (size_t)blockIdx.y * n * n;
    float mx = -3.4e38f;
    for (int i = 0; i < n; ++i) {
        float x = Sh[i * n + m];
        mx = x > mx ? x : mx;
    }
    float sum = 0.f;
    for (int i = 0; i < n; ++i) {
        float e = __expf(Sh[i * n + m] - mx);
        Sh[i * n + m] = e;
        sum += e;
    }
    float inv = 1.0f / sum;
    for (int i = 0; i < n; ++i)
        Ah[i * n + m] = (_Float16)(Sh[i * n + m] * inv);
}

__global__ __launch_bounds__(256)
void cvt_f32_f16(const float* __restrict__ s, _Float16* __restrict__ d, int n)
{
    int i = (blockIdx.x * 256 + threadIdx.x) * 4;
    if (i < n) { // n is a multiple of 4
        float4 x = *(const float4*)(s + i);
        v4h y = {(_Float16)x.x, (_Float16)x.y, (_Float16)x.z, (_Float16)x.w};
        *(v4h*)(d + i) = y;
    }
}

__global__ __launch_bounds__(256)
void copy_f32(const float* __restrict__ src, float* __restrict__ dst, int n)
{
    int i = blockIdx.x * 256 + threadIdx.x;
    if (i < n) dst[i] = src[i];
}

extern "C" void kernel_launch(void* const* d_in, const int* in_sizes, int n_in,
                              void* d_out, int out_size, void* d_ws, size_t ws_size,
                              hipStream_t stream)
{
    constexpr int D = 1024, N = 2048, H = 8, DK = 128, DFF = 4096, L = 4;
    const float LAM = 0.08838834764831845f; // 1/sqrt(128)

    const float* X  = (const float*)d_in[0];
    const float* Q  = (const float*)d_in[1];
    const float* Kw = (const float*)d_in[2];
    const float* V  = (const float*)d_in[3];
    const float* W1 = (const float*)d_in[4];
    const float* b1 = (const float*)d_in[5];
    const float* W2 = (const float*)d_in[6];
    const float* b2 = (const float*)d_in[7];

    float* Xc = (float*)d_out;           // running residual stream [D,N] fp32

    char* p = (char*)d_ws;
    float*    S    = (float*)p;    p += (size_t)H * N * N * 4;    // 128 MB
    _Float16* A16  = (_Float16*)p; p += (size_t)H * N * N * 2;    //  64 MB
    _Float16* Xh   = (_Float16*)p; p += (size_t)D * N * 2;        //   4 MB
    _Float16* QXh  = (_Float16*)p; p += (size_t)H * DK * N * 2;   //   4 MB
    _Float16* KXh  = (_Float16*)p; p += (size_t)H * DK * N * 2;   //   4 MB
    _Float16* VXh  = (_Float16*)p; p += (size_t)H * D * N * 2;    //  32 MB
    _Float16* Hh   = (_Float16*)p; p += (size_t)DFF * N * 2;      //  16 MB
    _Float16* Wq   = (_Float16*)p; p += (size_t)L * H * DK * D * 2;
    _Float16* Wk   = (_Float16*)p; p += (size_t)L * H * DK * D * 2;
    _Float16* Wv   = (_Float16*)p; p += (size_t)L * H * D * D * 2;
    _Float16* W1h  = (_Float16*)p; p += (size_t)L * DFF * D * 2;
    _Float16* W2h  = (_Float16*)p; p += (size_t)L * DFF * D * 2;

    dim3 blk(256);
    auto cvt = [&](const float* src, _Float16* dst, int n) {
        cvt_f32_f16<<<dim3(n / 4 / 256), blk, 0, stream>>>(src, dst, n);
    };
    cvt(Q,  Wq,  L * H * DK * D);
    cvt(Kw, Wk,  L * H * DK * D);
    cvt(V,  Wv,  L * H * D * D);
    cvt(W1, W1h, L * DFF * D);
    cvt(W2, W2h, L * DFF * D);
    copy_f32<<<dim3(D * N / 256), blk, 0, stream>>>(X, Xc, D * N);
    cvt(X, Xh, D * N);

    for (int l = 0; l < L; ++l) {
        const _Float16* Ql  = Wq  + l * H * DK * D;
        const _Float16* Kl  = Wk  + l * H * DK * D;
        const _Float16* Vl  = Wv  + l * H * D * D;
        const _Float16* W1l = W1h + l * DFF * D;
        const _Float16* W2l = W2h + l * DFF * D;
        const float*    b1l = b1  + l * DFF;
        const float*    b2l = b2  + l * D;

        // QX[h] = Q[h] @ X  -> f16
        gemm_wmma_f16<<<dim3(N / TN, DK / TM, H), blk, 0, stream>>>(
            Ql, D, DK * D, 0, Xh, N, 0, nullptr, QXh, N, DK * N,
            nullptr, nullptr, D, 1, 1.0f, 0);
        // KX[h] = K[h] @ X  -> f16
        gemm_wmma_f16<<<dim3(N / TN, DK / TM, H), blk, 0, stream>>>(
            Kl, D, DK * D, 0, Xh, N, 0, nullptr, KXh, N, DK * N,
            nullptr, nullptr, D, 1, 1.0f, 0);
        // S[h] = (QX[h]^T @ KX[h]) * LAM  -> f32
        gemm_wmma_f16<<<dim3(N / TN, N / TM, H), blk, 0, stream>>>(
            QXh, N, DK * N, 1, KXh, N, DK * N, S, nullptr, N, N * N,
            nullptr, nullptr, DK, 1, LAM, 0);
        // A[h] = softmax over rows per column  -> f16
        softmax_cols<<<dim3(N / 256, H), blk, 0, stream>>>(S, A16, N);
        // VX[h] = V[h] @ X  -> f16
        gemm_wmma_f16<<<dim3(N / TN, D / TM, H), blk, 0, stream>>>(
            Vl, D, D * D, 0, Xh, N, 0, nullptr, VXh, N, D * N,
            nullptr, nullptr, D, 1, 1.0f, 0);
        // X += sum_h VX[h] @ A[h]   (dual write: f32 Xc and f16 Xh)
        gemm_wmma_f16<<<dim3(N / TN, D / TM, 1), blk, 0, stream>>>(
            VXh, N, D * N, 0, A16, N, N * N, Xc, Xh, N, 0,
            Xc, nullptr, N, H, 1.0f, 0);
        // Hh = relu(W1 @ X + b1)  -> f16
        gemm_wmma_f16<<<dim3(N / TN, DFF / TM, 1), blk, 0, stream>>>(
            W1l, D, 0, 0, Xh, N, 0, nullptr, Hh, N, 0,
            nullptr, b1l, D, 1, 1.0f, 1);
        // X += W2 @ Hh + b2   (dual write)
        gemm_wmma_f16<<<dim3(N / TN, D / TM, 1), blk, 0, stream>>>(
            W2l, DFF, 0, 0, Hh, N, 0, Xc, Xh, N, 0,
            Xc, b2l, DFF, 1, 1.0f, 0);
    }
    (void)in_sizes; (void)n_in; (void)out_size; (void)ws_size;
}
